// EdgeClassifier_83210696393430
// MI455X (gfx1250) — matrix operand order repaired and checked by reading, not compiled, not run on hardware
//
#include <hip/hip_runtime.h>
#include <hip/hip_bf16.h>

typedef __attribute__((ext_vector_type(16))) _Float16 v16h;
typedef __attribute__((ext_vector_type(8)))  _Float16 v8h;
typedef __attribute__((ext_vector_type(8)))  float    v8f;

#define N_NODES 16384
#define N_EDGES 131072
#define ED      10
#define IN1     16
#define H1      32
#define H2      16
#define KDIM    1024   // edge-MLP hidden (H1*H1)
#define WCOLS   512    // IN1*H1 == H1*H2
#define EPB     32     // edges per block (two WMMA M-tiles)
#define KPAD    32     // ED=10 padded to one WMMA K-step

// ---------------- prep kernels ----------------

__global__ void k_zero(float* __restrict__ p, int n) {
  int i = blockIdx.x * blockDim.x + threadIdx.x;
  if (i < n) p[i] = 0.f;
}

// W2: [KDIM][WCOLS] f32 -> W2t: [WCOLS][KDIM] f16 (transposed, WMMA B frags, GEMM2)
__global__ void k_prep_w2t(const float* __restrict__ W, _Float16* __restrict__ Wt) {
  int i = blockIdx.x * blockDim.x + threadIdx.x;   // over KDIM*WCOLS
  int k = i / WCOLS, n = i % WCOLS;
  Wt[n * KDIM + k] = (_Float16)W[k * WCOLS + n];
}

// W1: [ED][KDIM] f32 -> W1t: [KDIM][KPAD] f16 (transposed + zero-padded K, GEMM1)
__global__ void k_prep_w1t(const float* __restrict__ W, _Float16* __restrict__ Wt) {
  int i = blockIdx.x * blockDim.x + threadIdx.x;   // over KDIM*KPAD
  int n = i / KPAD, k = i % KPAD;
  Wt[n * KPAD + k] = (k < ED) ? (_Float16)W[k * KDIM + n] : (_Float16)0.f;
}

__global__ void k_count(const int* __restrict__ dst, float* __restrict__ cnt) {
  int e = blockIdx.x * blockDim.x + threadIdx.x;
  if (e < N_EDGES) atomicAdd(&cnt[dst[e]], 1.f);
}

// ---------------- fused NNConv edge kernel ----------------
// One workgroup = 32 edges (two 16-row WMMA M-tiles). Pipeline per tile:
//   1) h = relu(ea @ W1 + b1)   [32 x 1024]  WMMA, K=10 zero-padded to 32
//   2) w = h @ W2 + b2          [32 x 512]   WMMA over K=1024
//        (each wave owns 64 cols x both M-tiles -> every B frag feeds 2 wmma)
//   3) msg[e] = x[src[e]] @ w_e  per-edge [IN_C x OUT_C] matvec
//   4) atomic scatter-add into accum[dst]
template<int IN_C, int OUT_C>
__global__ __launch_bounds__(256)
void k_edge(const float*    __restrict__ ea,    // [E][ED]
            const _Float16* __restrict__ W1t,   // [KDIM][KPAD] f16 (transposed+padded)
            const float*    __restrict__ b1,    // [KDIM]
            const _Float16* __restrict__ W2t,   // [WCOLS][KDIM] f16 (transposed)
            const float*    __restrict__ b2,    // [WCOLS]
            const float*    __restrict__ xf,    // [N][IN_C]
            const int*      __restrict__ src,
            const int*      __restrict__ dst,
            float*          __restrict__ accum) // [N][OUT_C]
{
  // 64 KB region: first h (f16 [32][1024]), then reused as w (f32 [32][512])
  __shared__ __align__(16) unsigned char s_raw[EPB * KDIM * 2];
  _Float16* s_h = (_Float16*)s_raw;
  float*    s_w = (float*)s_raw;

  const int tid  = threadIdx.x;
  const int e0   = blockIdx.x * EPB;
  const int lane = tid & 31;
  const int wave = tid >> 5;                   // 0..7 (wave32)
  const int mrow = lane & 15;
  const int kA   = (lane >> 4) << 3;           // A frag: chunks at +0/+16 or +8/+24
  const int kB   = (lane >> 4) << 4;           // unused; B frags use kA chunks like A
  const int rbase = (lane >> 4) << 3;          // C frag: row offset 0 or 8

  // ---- GEMM1 (WMMA): h[32][1024] = relu(ea_pad[32][32] @ W1t^T + b1) ----
  // A-frag K pattern: elements 0..7 -> K=kA..kA+7 (live iff K<ED); 8..15 -> K>=16 (pad, 0)
  v16h ag0 = {};   // edge rows 0..15
  v16h ag1 = {};   // edge rows 16..31
  #pragma unroll
  for (int j = 0; j < 8; ++j) {
    int k = kA + j;
    float v0 = (k < ED) ? ea[(e0 + mrow) * ED + k]      : 0.f;
    float v1 = (k < ED) ? ea[(e0 + 16 + mrow) * ED + k] : 0.f;
    ag0[j] = (_Float16)v0;
    ag1[j] = (_Float16)v1;
  }

  // wave w produces n-tiles [8w, 8w+8) of h for both M-tiles (16 wmma/wave)
  #pragma unroll
  for (int t = 0; t < 8; ++t) {
    int nt = wave * 8 + t;
    const _Float16* pb = W1t + (size_t)(nt * 16 + mrow) * KPAD + kA;
    v8h blo = *(const v8h*)(pb);
    v8h bhi = *(const v8h*)(pb + 16);
    v16h bf = __builtin_shufflevector(blo, bhi,
        0,1,2,3,4,5,6,7,8,9,10,11,12,13,14,15);
    v8f zf = {};
    v8f c0 = __builtin_amdgcn_wmma_f32_16x16x32_f16(
        false, ag0, false, bf, (short)0, zf, false, false);
    v8f c1 = __builtin_amdgcn_wmma_f32_16x16x32_f16(
        false, ag1, false, bf, (short)0, zf, false, false);
    int col = nt * 16 + mrow;
    float bb = b1[col];
    #pragma unroll
    for (int r = 0; r < 8; ++r) {
      int row = rbase + r;
      float x0 = c0[r] + bb, x1 = c1[r] + bb;
      s_h[row * KDIM + col]        = (_Float16)(x0 > 0.f ? x0 : 0.f);
      s_h[(16 + row) * KDIM + col] = (_Float16)(x1 > 0.f ? x1 : 0.f);
    }
  }
  __syncthreads();

  // ---- GEMM2 (WMMA): [32 x 1024] x [1024 x 512]; wave owns 64 cols x 2 M-tiles ----
  const int nb = wave * 64;

  v8f zf = {};
  v8f acc0[4] = { zf, zf, zf, zf };            // M-rows 0..15
  v8f acc1[4] = { zf, zf, zf, zf };            // M-rows 16..31

  for (int k0 = 0; k0 < KDIM; k0 += 32) {
    // A fragments (16-bit A 16x32: lanes 0-15 K=0..7,16..23; lanes 16-31 K=8..15,24..31)
    const _Float16* pa0 = s_h + mrow * KDIM + k0 + kA;
    const _Float16* pa1 = pa0 + 16 * KDIM;
    v8h a0lo = *(const v8h*)(pa0);
    v8h a0hi = *(const v8h*)(pa0 + 16);
    v8h a1lo = *(const v8h*)(pa1);
    v8h a1hi = *(const v8h*)(pa1 + 16);
    v16h af0 = __builtin_shufflevector(a0lo, a0hi,
        0,1,2,3,4,5,6,7,8,9,10,11,12,13,14,15);
    v16h af1 = __builtin_shufflevector(a1lo, a1hi,
        0,1,2,3,4,5,6,7,8,9,10,11,12,13,14,15);

    #pragma unroll
    for (int t = 0; t < 4; ++t) {
      // B fragment (32x16 B: lane%16 = col N; lanes<16 K=0..7,16..23; lanes>=16 K=8..15,24..31)
      const _Float16* pb = W2t + (size_t)(nb + 16 * t + mrow) * KDIM + k0 + kA;
      if (t == 0 && k0 + 32 < KDIM)
        __builtin_prefetch(pb + 32, 0, 1);     // global_prefetch_b8 of next K tile
      v8h blo = *(const v8h*)(pb);
      v8h bhi = *(const v8h*)(pb + 16);
      v16h bf = __builtin_shufflevector(blo, bhi,
          0,1,2,3,4,5,6,7,8,9,10,11,12,13,14,15);
      acc0[t] = __builtin_amdgcn_wmma_f32_16x16x32_f16(
          false, af0, false, bf, (short)0, acc0[t], false, false);
      acc1[t] = __builtin_amdgcn_wmma_f32_16x16x32_f16(
          false, af1, false, bf, (short)0, acc1[t], false, false);
    }
  }
  __syncthreads();   // all waves done reading s_h; reuse region as f32 w-tile

  // C/D layout: VGPR r -> row (rbase + r), col = nb + 16*t + lane%16
  #pragma unroll
  for (int t = 0; t < 4; ++t) {
    int col = nb + 16 * t + mrow;
    float bb = b2[col];
    #pragma unroll
    for (int r = 0; r < 8; ++r) {
      int row = rbase + r;
      s_w[row * WCOLS + col]        = acc0[t][r] + bb;
      s_w[(16 + row) * WCOLS + col] = acc1[t][r] + bb;
    }
  }
  __syncthreads();

  // ---- per-edge message + scatter-add:  msg[m][o] = sum_i x[src][i] * w[m][i*OUT_C + o]
  for (int idx = tid; idx < EPB * OUT_C; idx += 256) {
    int m = idx / OUT_C, o = idx % OUT_C;
    int e = e0 + m;
    int s = src[e];
    float v = 0.f;
    #pragma unroll
    for (int i = 0; i < IN_C; ++i)
      v += xf[s * IN_C + i] * s_w[m * WCOLS + i * OUT_C + o];
    atomicAdd(&accum[dst[e] * OUT_C + o], v);
  }
}

// ---------------- node update: mean + root + bias, relu ----------------
template<int IN_C, int OUT_C>
__global__ void k_node(const float* __restrict__ sum, const float* __restrict__ cnt,
                       const float* __restrict__ xf, const float* __restrict__ root,
                       const float* __restrict__ bias, float* __restrict__ out) {
  int idx = blockIdx.x * blockDim.x + threadIdx.x;
  if (idx >= N_NODES * OUT_C) return;
  int n = idx / OUT_C, o = idx % OUT_C;
  float c = cnt[n]; c = c > 1.f ? c : 1.f;
  float v = sum[idx] / c + bias[o];
  #pragma unroll
  for (int i = 0; i < IN_C; ++i) v += xf[n * IN_C + i] * root[i * OUT_C + o];
  out[idx] = v > 0.f ? v : 0.f;
}

// ---------------- final edge classifier: sigmoid(concat(h[src],h[dst]) @ fcW + b) ----------------
__global__ void k_final(const float* __restrict__ h, const int* __restrict__ src,
                        const int* __restrict__ dst, const float* __restrict__ fcW,
                        const float* __restrict__ fcb, float* __restrict__ out) {
  int e = blockIdx.x * blockDim.x + threadIdx.x;
  if (e >= N_EDGES) return;
  int s = src[e], d = dst[e];
  float v = fcb[0];
  #pragma unroll
  for (int j = 0; j < H2; ++j) v += h[s * H2 + j] * fcW[j];
  #pragma unroll
  for (int j = 0; j < H2; ++j) v += h[d * H2 + j] * fcW[H2 + j];
  out[e] = 1.f / (1.f + __expf(-v));
}

// ---------------- launch ----------------
extern "C" void kernel_launch(void* const* d_in, const int* in_sizes, int n_in,
                              void* d_out, int out_size, void* d_ws, size_t ws_size,
                              hipStream_t stream) {
  (void)in_sizes; (void)n_in; (void)out_size; (void)ws_size;

  const float* x    = (const float*)d_in[0];
  const int*   ei   = (const int*)  d_in[1];
  const float* eatt = (const float*)d_in[2];
  const float* l1W1 = (const float*)d_in[3];
  const float* l1b1 = (const float*)d_in[4];
  const float* l1W2 = (const float*)d_in[5];
  const float* l1b2 = (const float*)d_in[6];
  const float* l1rt = (const float*)d_in[7];
  const float* l1bs = (const float*)d_in[8];
  const float* l2W1 = (const float*)d_in[9];
  const float* l2b1 = (const float*)d_in[10];
  const float* l2W2 = (const float*)d_in[11];
  const float* l2b2 = (const float*)d_in[12];
  const float* l2rt = (const float*)d_in[13];
  const float* l2bs = (const float*)d_in[14];
  const float* fcW  = (const float*)d_in[15];
  const float* fcb  = (const float*)d_in[16];
  const int* src = ei;
  const int* dst = ei + N_EDGES;
  float* out = (float*)d_out;

  // workspace layout (~8.2 MB)
  char* ws = (char*)d_ws;
  _Float16* w2t1 = (_Float16*)ws; ws += (size_t)WCOLS * KDIM * sizeof(_Float16);
  _Float16* w2t2 = (_Float16*)ws; ws += (size_t)WCOLS * KDIM * sizeof(_Float16);
  _Float16* w1t1 = (_Float16*)ws; ws += (size_t)KDIM * KPAD * sizeof(_Float16);
  _Float16* w1t2 = (_Float16*)ws; ws += (size_t)KDIM * KPAD * sizeof(_Float16);
  float* sum1 = (float*)ws; ws += (size_t)N_NODES * H1 * sizeof(float);
  float* sum2 = (float*)ws; ws += (size_t)N_NODES * H2 * sizeof(float);
  float* cnt  = (float*)ws; ws += (size_t)N_NODES * sizeof(float);
  float* h1n  = (float*)ws; ws += (size_t)N_NODES * H1 * sizeof(float);
  float* h2n  = (float*)ws; ws += (size_t)N_NODES * H2 * sizeof(float);

  // zero accumulators (sum1|sum2|cnt are contiguous)
  int nz = N_NODES * (H1 + H2 + 1);
  k_zero<<<(nz + 255) / 256, 256, 0, stream>>>(sum1, nz);

  // weight prep: transpose/convert W2 (f16), transpose/pad W1 (f16)
  k_prep_w2t<<<(KDIM * WCOLS + 255) / 256, 256, 0, stream>>>(l1W2, w2t1);
  k_prep_w2t<<<(KDIM * WCOLS + 255) / 256, 256, 0, stream>>>(l2W2, w2t2);
  k_prep_w1t<<<(KDIM * KPAD + 255) / 256, 256, 0, stream>>>(l1W1, w1t1);
  k_prep_w1t<<<(KDIM * KPAD + 255) / 256, 256, 0, stream>>>(l2W1, w1t2);

  // in-degree counts (shared by both layers)
  k_count<<<N_EDGES / 256, 256, 0, stream>>>(dst, cnt);

  // layer 1: NNConv(16 -> 32)
  k_edge<IN1, H1><<<N_EDGES / EPB, 256, 0, stream>>>(eatt, w1t1, l1b1, w2t1, l1b2,
                                                     x, src, dst, sum1);
  k_node<IN1, H1><<<(N_NODES * H1 + 255) / 256, 256, 0, stream>>>(sum1, cnt, x,
                                                                  l1rt, l1bs, h1n);

  // layer 2: NNConv(32 -> 16)
  k_edge<H1, H2><<<N_EDGES / EPB, 256, 0, stream>>>(eatt, w1t2, l2b1, w2t2, l2b2,
                                                    h1n, src, dst, sum2);
  k_node<H1, H2><<<(N_NODES * H2 + 255) / 256, 256, 0, stream>>>(sum2, cnt, h1n,
                                                                 l2rt, l2bs, h2n);

  // final edge classifier
  k_final<<<N_EDGES / 256, 256, 0, stream>>>(h2n, src, dst, fcW, fcb, out);
}
